// Classifier_3616362463717
// MI455X (gfx1250) — matrix operand order repaired and checked by reading, not compiled, not run on hardware
//
#include <hip/hip_runtime.h>
#include <hip/hip_bf16.h>

// ============================================================================
// Fused ParT-style classifier for MI455X (gfx1250, wave32, WMMA + TDM).
// One workgroup (256 threads = 8 wave32 waves) per batch event; activations
// LDS-resident; dense math on v_wmma_f32_16x16x32_f16 with weights pre-packed
// into f16 B-fragment order; event input staged by the Tensor Data Mover.
// ============================================================================

typedef __attribute__((ext_vector_type(16))) _Float16 v16h;
typedef __attribute__((ext_vector_type(8)))  float    v8f;
typedef __attribute__((ext_vector_type(4)))  unsigned v4u;
typedef __attribute__((ext_vector_type(8)))  int      v8i;
typedef __attribute__((ext_vector_type(4)))  int      v4i;

#define BATCH 512
#define NP    132          // particles
#define NR    133          // particles + class token
#define NRP   144          // padded rows (9 full 16-row M tiles)
#define FDIM  13
#define PD    64
#define HEADS 4
#define DK    16
#define KNN   10
#define NPAIR (NP*KNN)
#define NTH   256
#define NWV   8
#define S     68           // padded row stride (floats) for big LDS buffers
#define HS    132          // padded stride for edge-MLP hidden tile (128+4)

// ---- input pointer table (setup_inputs() dict order, params flattened) ----
enum {
  I_part = 0, I_evt, I_points, I_mask,
  I_W_evt1, I_b_evt1, I_W_evt2, I_b_evt2,
  I_W_part, I_b_part,
  I_W_l1_0, I_b_l1_0, I_W_l2_0, I_b_l2_0, I_g_lgn_0, I_b_lgn_0,
  I_W_l1_1, I_b_l1_1, I_W_l2_1, I_b_l2_1, I_g_lgn_1, I_b_lgn_1,
  I_ls_local, I_g_t1, I_b_t1,
  I_W_tq, I_b_tq, I_W_tk, I_b_tk, I_W_tv, I_b_tv, I_W_to, I_b_to,
  I_g_t2, I_b_t2, I_ls_attn, I_g_t3, I_b_t3,
  I_W_tf1, I_b_tf1, I_W_tf2, I_b_tf2, I_ls_ff, I_class_token,
  I_g_ca_0, I_b_ca_0, I_W_cq_0, I_b_cq_0, I_W_ck_0, I_b_ck_0,
  I_W_cv_0, I_b_cv_0, I_W_co_0, I_b_co_0,
  I_g_cb_0, I_b_cb_0, I_ls_ca_0, I_g_cc_0, I_b_cc_0,
  I_W_cf1_0, I_b_cf1_0, I_W_cf2_0, I_b_cf2_0, I_ls_cf_0,
  I_g_ca_1, I_b_ca_1, I_W_cq_1, I_b_cq_1, I_W_ck_1, I_b_ck_1,
  I_W_cv_1, I_b_cv_1, I_W_co_1, I_b_co_1,
  I_g_cb_1, I_b_cb_1, I_ls_ca_1, I_g_cc_1, I_b_cc_1,
  I_W_cf1_1, I_b_cf1_1, I_W_cf2_1, I_b_cf2_1, I_ls_cf_1,
  I_g_out, I_b_out, I_W_cls, I_b_cls,
  I_COUNT
};

struct Params { const float* p[I_COUNT]; };

// ---- pre-packed f16 weight fragment offsets (halves) ----------------------
constexpr int PK_part = 0;                      // Kp=32,  N=64
constexpr int PK_l1_0 = PK_part + 32 * 64;      // Kp=32,  N=128
constexpr int PK_l2_0 = PK_l1_0 + 32 * 128;     // Kp=128, N=64
constexpr int PK_l1_1 = PK_l2_0 + 128 * 64;     // Kp=64,  N=128
constexpr int PK_l2_1 = PK_l1_1 + 64 * 128;     // Kp=128, N=64
constexpr int PK_tq   = PK_l2_1 + 128 * 64;
constexpr int PK_tk   = PK_tq + 64 * 64;
constexpr int PK_tv   = PK_tk + 64 * 64;
constexpr int PK_to   = PK_tv + 64 * 64;
constexpr int PK_tf1  = PK_to + 64 * 64;        // Kp=64,  N=256
constexpr int PK_tf2  = PK_tf1 + 64 * 256;      // Kp=256, N=64
constexpr int PK_ck0  = PK_tf2 + 256 * 64;
constexpr int PK_cv0  = PK_ck0 + 64 * 64;
constexpr int PK_ck1  = PK_cv0 + 64 * 64;
constexpr int PK_cv1  = PK_ck1 + 64 * 64;
constexpr int PK_TOTAL= PK_cv1 + 64 * 64;       // 96256 halves = 192512 B

constexpr int SMEM_WORDS =
    5*NRP*S + 16*S + 16*HS + 2*NTH + NP + NP + NP*2 + 128 + 6*PD + 256
  + NPAIR + 16;

__device__ __forceinline__ float gelu_f(float x) {
  return 0.5f * x * (1.0f + erff(x * 0.70710678118654752f));
}

// ---------------------------------------------------------------------------
// TDM: 1D DMA of `ndw` dwords global -> LDS (D# per ISA 08_async_tensor 8.3/8.4:
// count=1, type=2, data_size=4B, 1-row 2D tile, no pad/iterate/multicast).
// ---------------------------------------------------------------------------
__device__ static inline void tdm_load_dwords(const float* gsrc, float* ldst, int ndw)
{
  const unsigned lds_addr = (unsigned)(unsigned long long)ldst;
  const unsigned long long ga = (unsigned long long)gsrc;
  v4u g0;
  g0[0] = 1u;                                          // count=1, user mode
  g0[1] = lds_addr;                                    // lds_addr [63:32]
  g0[2] = (unsigned)(ga & 0xFFFFFFFFu);                // global_addr lo
  g0[3] = (unsigned)((ga >> 32) & 0x01FFFFFFu) | 0x80000000u; // hi25 | type=2
  v8i g1;
  g1[0] = 0x00020000;                                  // data_size=4B, mask=0
  g1[1] = (int)((unsigned)(ndw & 0xFFFF) << 16);       // tensor_dim0 lo16
  g1[2] = (int)((((unsigned)ndw >> 16) & 0xFFFFu) | (1u << 16)); // dim0 hi16, tensor_dim1=1
  g1[3] = (int)((unsigned)(ndw & 0xFFFF) << 16);       // tile_dim0
  g1[4] = 1;                                           // tile_dim1=1
  g1[5] = ndw;                                         // tensor_dim0_stride lo32
  g1[6] = (int)((unsigned)(ndw & 0xFFFF) << 16);       // dim1_stride lo16
  g1[7] = 0;
  const v4i gz = {0, 0, 0, 0};
#if __clang_major__ >= 23
  const v8i gz8 = {0, 0, 0, 0, 0, 0, 0, 0};
  __builtin_amdgcn_tensor_load_to_lds(g0, g1, gz, gz, gz8, 0);
#else
  __builtin_amdgcn_tensor_load_to_lds(g0, g1, gz, gz, 0);
#endif
}

// ---------------------------------------------------------------------------
// One-time weight pre-pack: f32 row-major -> f16 WMMA B-fragment order in ws.
// ---------------------------------------------------------------------------
extern "C" __global__ void __launch_bounds__(NTH)
prepack_weights_kernel(Params pp, _Float16* __restrict__ wp)
{
  constexpr int NM = 15;
  const int srcs[NM] = {I_W_part,I_W_l1_0,I_W_l2_0,I_W_l1_1,I_W_l2_1,
                        I_W_tq,I_W_tk,I_W_tv,I_W_to,I_W_tf1,I_W_tf2,
                        I_W_ck_0,I_W_cv_0,I_W_ck_1,I_W_cv_1};
  const int Ks[NM]   = {13,13,128,64,128,64,64,64,64,64,256,64,64,64,64};
  const int Ns[NM]   = {64,128,64,128,64,64,64,64,64,256,64,64,64,64,64};
  const int offs[NM+1] = {PK_part,PK_l1_0,PK_l2_0,PK_l1_1,PK_l2_1,PK_tq,PK_tk,
                          PK_tv,PK_to,PK_tf1,PK_tf2,PK_ck0,PK_cv0,PK_ck1,PK_cv1,
                          PK_TOTAL};
  for (int g = blockIdx.x * NTH + threadIdx.x; g < PK_TOTAL; g += gridDim.x * NTH) {
    int mi = 0;
    while (offs[mi + 1] <= g) ++mi;
    const int local  = g - offs[mi];
    const int NT     = Ns[mi] >> 4;
    const int tile   = local >> 9;
    const int within = local & 511;
    const int lane   = within >> 4;
    const int e      = within & 15;
    const int kt     = tile / NT, nt = tile - kt * NT;
    const int k      = (kt << 5) + ((e >> 3) << 4) + ((lane >> 4) << 3) + (e & 7);
    const int c      = (nt << 4) + (lane & 15);
    float v = 0.f;
    if (k < Ks[mi]) v = pp.p[srcs[mi]][k * Ns[mi] + c];
    wp[g] = (_Float16)v;
  }
}

// ---------------------------------------------------------------------------
// Wave-cooperative WMMA GEMM on pre-packed weights; compile-time epilogue.
// Full M-tiles take the unguarded fast path; only the last ragged tile guards.
// ---------------------------------------------------------------------------
enum { FG_GELU = 1, FG_ACC = 2, FG_BIAS = 4, FG_ADDV = 8, FG_RSC = 16 };

template<int F>
__device__ static inline void gemm_p(
    const float* A, int lda, int M, int Kp,
    const _Float16* __restrict__ Wp, int wNT, int kt0, int nt0,
    const float* __restrict__ bias,
    const float* __restrict__ addv,
    const float* rowscale,
    float* O, int ldo, int N)
{
  const int lane = threadIdx.x & 31;
  const int wave = threadIdx.x >> 5;
  const int mt = (M + 15) >> 4, nt = N >> 4;
  const int hq = (lane >> 4) << 3;
  const int rr = lane & 15;
  for (int t = wave; t < mt * nt; t += NWV) {
    const int mi = t / nt, ni = t - mi * nt;
    const int m0 = mi << 4, n0 = ni << 4;
    v8f acc = {0.f,0.f,0.f,0.f,0.f,0.f,0.f,0.f};
    const float* arow = A + (m0 + rr) * lda + hq;
    const _Float16* wtile = Wp + (((size_t)kt0 * wNT + (nt0 + ni)) * 32 + lane) * 16;
    for (int kt = 0; kt < (Kp >> 5); ++kt) {
      float ta[16];
      const float* ap = arow + (kt << 5);
      *(float4*)(ta)      = *(const float4*)(ap);       // ds_load_b128
      *(float4*)(ta + 4)  = *(const float4*)(ap + 4);
      *(float4*)(ta + 8)  = *(const float4*)(ap + 16);
      *(float4*)(ta + 12) = *(const float4*)(ap + 20);
      v16h af;
      #pragma unroll
      for (int e = 0; e < 16; ++e) af[e] = (_Float16)ta[e];
      const v16h bf = *(const v16h*)(wtile + (size_t)kt * (wNT << 9));
      acc = __builtin_amdgcn_wmma_f32_16x16x32_f16(
          false, af, false, bf, (short)0, acc, false, false);
    }
    const int c = n0 + (lane & 15);
    float bcol = 0.f;
    if (F & FG_BIAS) bcol += bias[c];
    if (F & FG_ADDV) bcol += addv[c];
    if (m0 + 16 <= M) {                       // fast path: full tile
      float rsv[8];
      if (F & FG_RSC) {
        *(float4*)(rsv)     = *(const float4*)(rowscale + m0 + hq);
        *(float4*)(rsv + 4) = *(const float4*)(rowscale + m0 + hq + 4);
      }
      #pragma unroll
      for (int v = 0; v < 8; ++v) {
        const int r = m0 + v + hq;
        float val = acc[v] + bcol;
        if (F & FG_GELU) val = gelu_f(val);
        if (F & FG_RSC)  val *= rsv[v];
        if (F & FG_ACC)  O[r * ldo + c] += val;
        else             O[r * ldo + c]  = val;
      }
    } else {                                  // ragged last tile
      #pragma unroll
      for (int v = 0; v < 8; ++v) {
        const int r = m0 + v + hq;
        if (r < M) {
          float val = acc[v] + bcol;
          if (F & FG_GELU) val = gelu_f(val);
          if (F & FG_RSC)  val *= rowscale[r];
          if (F & FG_ACC)  O[r * ldo + c] += val;
          else             O[r * ldo + c]  = val;
        }
      }
    }
  }
}

// Block-wide mean / rsqrt(var+eps) over rows x 64 cols of an LDS matrix.
__device__ static inline void block_stats(const float* X, int ld, int rows,
                                          float* red, float& mu, float& rs)
{
  const int tid = threadIdx.x;
  float s = 0.f, ss = 0.f;
  for (int i = tid; i < rows * PD; i += NTH) {
    const float v = X[(i >> 6) * ld + (i & 63)];
    s += v; ss += v * v;
  }
  red[tid] = s; red[NTH + tid] = ss;
  __syncthreads();
  for (int off = NTH >> 1; off > 0; off >>= 1) {
    if (tid < off) { red[tid] += red[tid + off]; red[NTH + tid] += red[NTH + tid + off]; }
    __syncthreads();
  }
  const float inv = 1.f / (float)(rows * PD);
  mu = red[0] * inv;
  const float var = red[NTH] * inv - mu * mu;
  rs = rsqrtf(var + 1.0e-3f);
  __syncthreads();
}

// Per-row KNN (10 nearest, self excluded); own row cached in registers.
template<int DIM>
__device__ static inline void knn_topk(const float* X, int ld,
                                       const float* shf, int* idxb, int tid)
{
  if (tid < NP) {
    const int n = tid;
    const float sn = shf[n];
    float xn[DIM];
    #pragma unroll
    for (int c = 0; c < DIM; ++c) xn[c] = X[n * ld + c] + sn;
    float bd[KNN]; int bi[KNN];
    #pragma unroll
    for (int k = 0; k < KNN; ++k) { bd[k] = 3.0e38f; bi[k] = 0; }
    for (int m = 0; m < NP; ++m) {
      if (m == n) continue;
      const float sm = shf[m];
      float d = 0.f;
      for (int c = 0; c < DIM; ++c) {
        const float t = xn[c] - (X[m * ld + c] + sm);
        d += t * t;
      }
      if (d < bd[KNN - 1]) {
        int k = KNN - 1;
        while (k > 0 && bd[k - 1] > d) { bd[k] = bd[k - 1]; bi[k] = bi[k - 1]; --k; }
        bd[k] = d; bi[k] = m;
      }
    }
    for (int k = 0; k < KNN; ++k) idxb[n * KNN + k] = bi[k];
  }
  __syncthreads();
}

// Second edge-MLP GEMM with ds_add_f32 scatter-accumulate.
__device__ static inline void edge_second_gemm(const float* Hed,
                                               const _Float16* __restrict__ Wp,
                                               float* Fout, const int* rowN,
                                               int ch, int tid)
{
  const int lane = tid & 31;
  const int wave = tid >> 5;
  const int hq = (lane >> 4) << 3;
  const int rr = lane & 15;
  for (int t = wave; t < 4; t += NWV) {
    const int n0 = t << 4;
    v8f acc = {0.f,0.f,0.f,0.f,0.f,0.f,0.f,0.f};
    const float* arow = Hed + rr * HS + hq;
    const _Float16* wtile = Wp + ((size_t)t * 32 + lane) * 16;
    #pragma unroll
    for (int kt = 0; kt < 4; ++kt) {
      float ta[16];
      const float* ap = arow + (kt << 5);
      *(float4*)(ta)      = *(const float4*)(ap);
      *(float4*)(ta + 4)  = *(const float4*)(ap + 4);
      *(float4*)(ta + 8)  = *(const float4*)(ap + 16);
      *(float4*)(ta + 12) = *(const float4*)(ap + 20);
      v16h af;
      #pragma unroll
      for (int e = 0; e < 16; ++e) af[e] = (_Float16)ta[e];
      const v16h bf = *(const v16h*)(wtile + (size_t)kt * (4 << 9));
      acc = __builtin_amdgcn_wmma_f32_16x16x32_f16(
          false, af, false, bf, (short)0, acc, false, false);
    }
    #pragma unroll
    for (int v = 0; v < 8; ++v) {
      const int j = v + hq;
      const int p = ch * 16 + j;
      if (p < NPAIR)
        atomicAdd(&Fout[rowN[j] * S + n0 + (lane & 15)], 0.1f * acc[v]);
    }
  }
}

// Stage 16 gathered edge rows (knn - center), K zero-padded; DINP is a
// compile-time constant so the index math is shifts, not division.
template<int DINP, int DIN>
__device__ static inline void stage_edge_tile(const float* Fin, const int* idxb,
                                              float* Aed, int ch, int tid)
{
  for (int e = tid; e < 16 * DINP; e += NTH) {
    const int j = e / DINP, c = e - j * DINP;
    const int p = ch * 16 + j;
    float v = 0.f;
    if (p < NPAIR && c < DIN) {
      const int n = p / KNN, m = idxb[p];
      v = Fin[m * S + c] - Fin[n * S + c];
    }
    Aed[j * S + c] = v;
  }
}

// One EdgeConv iteration (KNN -> gathered MLP -> mean -> groupnorm).
template<int DIN, int DINP, int CDIM>
__device__ static inline void edgeconv_iter(
    const float* coords, int cld, const float* Fin, float* Fout,
    const _Float16* W1p, const _Float16* W2p,
    const float* b1, const float* b2, const float* gg, const float* gb,
    const float* shf, int* idxb, int* rowN,
    float* Aed, float* Hed, float* red, int tid)
{
  knn_topk<CDIM>(coords, cld, shf, idxb, tid);
  for (int i = tid; i < NP * PD; i += NTH)
    Fout[(i >> 6) * S + (i & 63)] = b2[i & 63];
  __syncthreads();
  const int nchunk = (NPAIR + 15) / 16;
  for (int ch = 0; ch < nchunk; ++ch) {
    stage_edge_tile<DINP, DIN>(Fin, idxb, Aed, ch, tid);
    if (tid < 16) {
      const int p = ch * 16 + tid;
      rowN[tid] = (p < NPAIR) ? p / KNN : 0;
    }
    __syncthreads();
    gemm_p<FG_BIAS | FG_GELU>(Aed, S, 16, DINP, W1p, 8, 0, 0,
        b1, nullptr, nullptr, Hed, HS, 128);
    __syncthreads();
    edge_second_gemm(Hed, W2p, Fout, rowN, ch, tid);
    __syncthreads();
  }
  float mu, rs;
  block_stats(Fout, S, NP, red, mu, rs);
  for (int i = tid; i < NP * PD; i += NTH) {
    const int r = i >> 6, c = i & 63;
    Fout[r * S + c] = gg[c] * (Fout[r * S + c] - mu) * rs + gb[c];
  }
  __syncthreads();
}

// ============================================================================
extern "C" __global__ void __launch_bounds__(NTH)
classifier_fused_kernel(Params pp, const _Float16* __restrict__ wp,
                        float* __restrict__ out)
{
  const int b   = blockIdx.x;
  const int tid = threadIdx.x;
  extern __shared__ float smem[];

  float* xenc  = smem;             // residual / encoded  (NRP x S)
  float* xa    = xenc  + NRP * S;
  float* xb    = xa    + NRP * S;
  float* xc    = xb    + NRP * S;
  float* xd    = xc    + NRP * S;
  float* Aed   = xd    + NRP * S;  // 16 x S edge tile
  float* Hed   = Aed   + 16 * S;   // 16 x HS hidden tile (also TDM scratch)
  float* red   = Hed   + 16 * HS;
  float* maskp = red   + 2 * NTH;
  float* shf   = maskp + NP;
  float* pts2  = shf   + NP;
  float* evh   = pts2  + NP * 2;
  float* evenc = evh   + 128;
  float* tokv  = evenc + PD;
  float* ta    = tokv  + PD;
  float* tb    = ta    + PD;
  float* qrow  = tb    + PD;
  float* updv  = qrow  + PD;
  float* hsm   = updv  + PD;
  int*   idxb  = (int*)(hsm + 256);
  int*   rowN  = idxb + NPAIR;

  const float* gpart = pp.p[I_part]   + (size_t)b * NP * FDIM;
  const float* gpts  = pp.p[I_points] + (size_t)b * NP * 2;
  const float* gmask = pp.p[I_mask]   + (size_t)b * NP;
  const float* gevt  = pp.p[I_evt]    + (size_t)b * 2;

  // ---- TDM: DMA the event's particle block into LDS scratch ---------------
  if (tid == 0) {
    tdm_load_dwords(gpart, Hed, NP * FDIM);          // tensor_load_to_lds
    __builtin_prefetch(wp, 0, 0);                    // global_prefetch_b8
    __builtin_prefetch(wp + PK_tf1, 0, 0);
    __builtin_prefetch(wp + PK_tf2, 0, 0);
  }
  for (int i = tid; i < NP * 2; i += NTH) pts2[i] = gpts[i];
  for (int i = tid; i < NP; i += NTH) {
    const float m = gmask[i];
    maskp[i] = m;
    shf[i]   = (m == 0.f) ? 999.f : 0.f;
  }
  for (int i = tid; i < PD; i += NTH) tokv[i] = pp.p[I_class_token][i];
  __builtin_amdgcn_s_wait_tensorcnt(0);              // wave0 drains its TDM op
  __syncthreads();
  // spread compact part rows into padded xa (K zero-padded to 32)
  for (int i = tid; i < NP * 32; i += NTH) {
    const int r = i >> 5, c = i & 31;
    xa[r * S + c] = (c < FDIM) ? Hed[r * FDIM + c] : 0.f;
  }
  __syncthreads();

  // ---- event encoder: gelu(evt@W1+b1)@W2+b2 -------------------------------
  {
    const float e0 = gevt[0], e1 = gevt[1];
    if (tid < 128) {
      const float* W = pp.p[I_W_evt1];
      evh[tid] = gelu_f(pp.p[I_b_evt1][tid] + e0 * W[tid] + e1 * W[128 + tid]);
    }
    __syncthreads();
    if (tid < PD) {
      const float* W2 = pp.p[I_W_evt2];
      float acc = pp.p[I_b_evt2][tid];
      for (int j = 0; j < 128; ++j) acc += evh[j] * W2[j * PD + tid];
      evenc[tid] = acc;
    }
    __syncthreads();
  }

  // ---- particle embedding: encoded = (part@W_part + b + evt)*mask ---------
  gemm_p<FG_BIAS | FG_ADDV | FG_RSC>(xa, S, NP, 32, wp + PK_part, 4, 0, 0,
      pp.p[I_b_part], evenc, maskp, xenc, S, PD);
  __syncthreads();

  // ---- EdgeConv x2 --------------------------------------------------------
  edgeconv_iter<FDIM, 32, 2>(pts2, 2, xa, xb,
      wp + PK_l1_0, wp + PK_l2_0, pp.p[I_b_l1_0], pp.p[I_b_l2_0],
      pp.p[I_g_lgn_0], pp.p[I_b_lgn_0], shf, idxb, rowN, Aed, Hed, red, tid);
  edgeconv_iter<PD, PD, PD>(xb, S, xb, xc,
      wp + PK_l1_1, wp + PK_l2_1, pp.p[I_b_l1_1], pp.p[I_b_l2_1],
      pp.p[I_g_lgn_1], pp.p[I_b_lgn_1], shf, idxb, rowN, Aed, Hed, red, tid);

  // encoded += feats * ls_local * mask
  for (int i = tid; i < NP * PD; i += NTH) {
    const int r = i >> 6, c = i & 63;
    xenc[r * S + c] += xc[r * S + c] * pp.p[I_ls_local][c] * maskp[r];
  }
  __syncthreads();

  // ---- transformer block --------------------------------------------------
  float mu, rs;
  block_stats(xenc, S, NP, red, mu, rs);
  for (int i = tid; i < NP * PD; i += NTH) {
    const int r = i >> 6, c = i & 63;
    xa[r * S + c] = pp.p[I_g_t1][c] * (xenc[r * S + c] - mu) * rs + pp.p[I_b_t1][c];
  }
  __syncthreads();
  gemm_p<FG_BIAS>(xa, S, NP, PD, wp + PK_tq, 4, 0, 0, pp.p[I_b_tq], nullptr, nullptr, xb, S, PD);
  gemm_p<FG_BIAS>(xa, S, NP, PD, wp + PK_tk, 4, 0, 0, pp.p[I_b_tk], nullptr, nullptr, xc, S, PD);
  gemm_p<FG_BIAS>(xa, S, NP, PD, wp + PK_tv, 4, 0, 0, pp.p[I_b_tv], nullptr, nullptr, xd, S, PD);
  __syncthreads();

  // masked MHA, branchless online softmax, float4 K/V loads
  for (int task = tid; task < NP * HEADS; task += NTH) {
    const int n = task >> 2, h = task & 3;
    float qr[DK];
    *(float4*)(qr)      = *(const float4*)(&xb[n * S + h * DK]);
    *(float4*)(qr + 4)  = *(const float4*)(&xb[n * S + h * DK + 4]);
    *(float4*)(qr + 8)  = *(const float4*)(&xb[n * S + h * DK + 8]);
    *(float4*)(qr + 12) = *(const float4*)(&xb[n * S + h * DK + 12]);
    const float mn = maskp[n];
    float mx = -3.4e38f, l = 0.f, acc[DK];
    #pragma unroll
    for (int d = 0; d < DK; ++d) acc[d] = 0.f;
    for (int m = 0; m < NP; ++m) {
      float kv[DK];
      const float* kp = &xc[m * S + h * DK];
      *(float4*)(kv)      = *(const float4*)(kp);
      *(float4*)(kv + 4)  = *(const float4*)(kp + 4);
      *(float4*)(kv + 8)  = *(const float4*)(kp + 8);
      *(float4*)(kv + 12) = *(const float4*)(kp + 12);
      float sdot = 0.f;
      #pragma unroll
      for (int d = 0; d < DK; ++d) sdot += qr[d] * kv[d];
      const float sVal = (mn * maskp[m] > 0.f) ? sdot * 0.125f : -1.0e9f;
      const float nmx = fmaxf(mx, sVal);
      const float f = __expf(mx - nmx);
      const float w = __expf(sVal - nmx);
      mx = nmx;
      l = l * f + w;
      const float* vp = &xd[m * S + h * DK];
      *(float4*)(kv)      = *(const float4*)(vp);
      *(float4*)(kv + 4)  = *(const float4*)(vp + 4);
      *(float4*)(kv + 8)  = *(const float4*)(vp + 8);
      *(float4*)(kv + 12) = *(const float4*)(vp + 12);
      #pragma unroll
      for (int d = 0; d < DK; ++d) acc[d] = acc[d] * f + w * kv[d];
    }
    const float il = 1.f / l;
    #pragma unroll
    for (int d = 0; d < DK; ++d) xa[n * S + h * DK + d] = acc[d] * il;
  }
  __syncthreads();

  gemm_p<FG_BIAS>(xa, S, NP, PD, wp + PK_to, 4, 0, 0, pp.p[I_b_to], nullptr, nullptr, xb, S, PD);
  __syncthreads();
  block_stats(xb, S, NP, red, mu, rs);
  for (int i = tid; i < NP * PD; i += NTH) {
    const int r = i >> 6, c = i & 63;
    float u = (pp.p[I_g_t2][c] * (xb[r * S + c] - mu) * rs + pp.p[I_b_t2][c]) * maskp[r];
    u *= pp.p[I_ls_attn][c] * maskp[r];
    xenc[r * S + c] += u;
  }
  __syncthreads();
  block_stats(xenc, S, NP, red, mu, rs);
  for (int i = tid; i < NP * PD; i += NTH) {
    const int r = i >> 6, c = i & 63;
    xa[r * S + c] = pp.p[I_g_t3][c] * (xenc[r * S + c] - mu) * rs + pp.p[I_b_t3][c];
    xc[r * S + c] = pp.p[I_b_tf2][c];
  }
  __syncthreads();
  for (int c4 = 0; c4 < 4; ++c4) {
    gemm_p<FG_BIAS | FG_GELU>(xa, S, NP, PD, wp + PK_tf1, 16, 0, c4 * 4,
        pp.p[I_b_tf1] + c4 * 64, nullptr, nullptr, xb, S, 64);
    __syncthreads();
    gemm_p<FG_ACC>(xb, S, NP, 64, wp + PK_tf2, 4, c4 * 2, 0,
        nullptr, nullptr, nullptr, xc, S, PD);
    __syncthreads();
  }
  for (int i = tid; i < NP * PD; i += NTH) {
    const int r = i >> 6, c = i & 63;
    const float v = xc[r * S + c] * pp.p[I_ls_ff][c] * maskp[r];
    xenc[r * S + c] = (v + xenc[r * S + c]) * maskp[r];
  }
  __syncthreads();

  // ---- class-attention blocks x2 -----------------------------------------
  for (int j = 0; j < 2; ++j) {
    const int off = j * 20;
    const _Float16* Wkp = wp + (j ? PK_ck1 : PK_ck0);
    const _Float16* Wvp = wp + (j ? PK_cv1 : PK_cv0);
    for (int i = tid; i < NR * PD; i += NTH) {
      const int r = i >> 6, c = i & 63;
      xa[r * S + c] = (r == 0) ? tokv[c] : xenc[(r - 1) * S + c];
    }
    __syncthreads();
    block_stats(xa, S, NR, red, mu, rs);
    const float* g1  = pp.p[I_g_ca_0 + off];
    const float* bb1 = pp.p[I_b_ca_0 + off];
    for (int i = tid; i < NR * PD; i += NTH) {
      const int r = i >> 6, c = i & 63;
      xb[r * S + c] = g1[c] * (xa[r * S + c] - mu) * rs + bb1[c];
    }
    __syncthreads();
    if (tid < PD) {
      const float* Wq = pp.p[I_W_cq_0 + off];
      float acc = pp.p[I_b_cq_0 + off][tid];
      for (int d = 0; d < PD; ++d) acc += xb[d] * Wq[d * PD + tid];
      qrow[tid] = acc;
    }
    gemm_p<FG_BIAS>(xb, S, NR, PD, Wkp, 4, 0, 0, pp.p[I_b_ck_0 + off], nullptr, nullptr, xc, S, PD);
    gemm_p<FG_BIAS>(xb, S, NR, PD, Wvp, 4, 0, 0, pp.p[I_b_cv_0 + off], nullptr, nullptr, xd, S, PD);
    __syncthreads();
    if (tid < HEADS) {
      const int h = tid;
      float mx2 = -3.4e38f, l2 = 0.f, acc[DK];
      #pragma unroll
      for (int d = 0; d < DK; ++d) acc[d] = 0.f;
      for (int m = 0; m < NR; ++m) {
        float kv[DK];
        const float* kp = &xc[m * S + h * DK];
        *(float4*)(kv)      = *(const float4*)(kp);
        *(float4*)(kv + 4)  = *(const float4*)(kp + 4);
        *(float4*)(kv + 8)  = *(const float4*)(kp + 8);
        *(float4*)(kv + 12) = *(const float4*)(kp + 12);
        float sdot = 0.f;
        #pragma unroll
        for (int d = 0; d < DK; ++d) sdot += qrow[h * DK + d] * kv[d];
        const float sVal = sdot * 0.125f;
        const float nmx = fmaxf(mx2, sVal);
        const float f = __expf(mx2 - nmx);
        const float w = __expf(sVal - nmx);
        mx2 = nmx;
        l2 = l2 * f + w;
        const float* vp = &xd[m * S + h * DK];
        *(float4*)(kv)      = *(const float4*)(vp);
        *(float4*)(kv + 4)  = *(const float4*)(vp + 4);
        *(float4*)(kv + 8)  = *(const float4*)(vp + 8);
        *(float4*)(kv + 12) = *(const float4*)(vp + 12);
        #pragma unroll
        for (int d = 0; d < DK; ++d) acc[d] = acc[d] * f + w * kv[d];
      }
      const float il = 1.f / l2;
      #pragma unroll
      for (int d = 0; d < DK; ++d) updv[h * DK + d] = acc[d] * il;
    }
    __syncthreads();
    if (tid < PD) {
      const float* Wo = pp.p[I_W_co_0 + off];
      float acc = pp.p[I_b_co_0 + off][tid];
      for (int d = 0; d < PD; ++d) acc += updv[d] * Wo[d * PD + tid];
      ta[tid] = acc;
    }
    __syncthreads();
    block_stats(ta, PD, 1, red, mu, rs);
    if (tid < PD)
      tb[tid] = (pp.p[I_g_cb_0 + off][tid] * (ta[tid] - mu) * rs +
                 pp.p[I_b_cb_0 + off][tid]) * pp.p[I_ls_ca_0 + off][tid] + tokv[tid];
    __syncthreads();
    block_stats(tb, PD, 1, red, mu, rs);
    if (tid < PD)
      ta[tid] = pp.p[I_g_cc_0 + off][tid] * (tb[tid] - mu) * rs + pp.p[I_b_cc_0 + off][tid];
    __syncthreads();
    {
      const float* W1 = pp.p[I_W_cf1_0 + off];
      float acc = pp.p[I_b_cf1_0 + off][tid];
      for (int d = 0; d < PD; ++d) acc += ta[d] * W1[d * 256 + tid];
      hsm[tid] = gelu_f(acc);
    }
    __syncthreads();
    if (tid < PD) {
      const float* W2 = pp.p[I_W_cf2_0 + off];
      float acc = pp.p[I_b_cf2_0 + off][tid];
      for (int k = 0; k < 256; ++k) acc += hsm[k] * W2[k * PD + tid];
      tokv[tid] = acc * pp.p[I_ls_cf_0 + off][tid] + tb[tid];
    }
    __syncthreads();
  }

  // ---- head: gn([tok;enc]) row0 @ W_cls -> sigmoid ------------------------
  for (int i = tid; i < NR * PD; i += NTH) {
    const int r = i >> 6, c = i & 63;
    xa[r * S + c] = (r == 0) ? tokv[c] : xenc[(r - 1) * S + c];
  }
  __syncthreads();
  block_stats(xa, S, NR, red, mu, rs);
  if (tid < PD)
    ta[tid] = pp.p[I_g_out][tid] * (xa[tid] - mu) * rs + pp.p[I_b_out][tid];
  __syncthreads();
  if (tid == 0) {
    float logit = pp.p[I_b_cls][0];
    for (int d = 0; d < PD; ++d) logit += ta[d] * pp.p[I_W_cls][d];
    out[b] = 1.f / (1.f + __expf(-logit));
  }
}

// ============================================================================
extern "C" void kernel_launch(void* const* d_in, const int* in_sizes, int n_in,
                              void* d_out, int out_size, void* d_ws, size_t ws_size,
                              hipStream_t stream)
{
  (void)in_sizes; (void)out_size; (void)ws_size;
  Params pp;
  for (int i = 0; i < I_COUNT; ++i)
    pp.p[i] = (i < n_in) ? (const float*)d_in[i] : nullptr;

  _Float16* wp = (_Float16*)d_ws;

  const int pk_blocks = (PK_TOTAL + NTH - 1) / NTH;
  prepack_weights_kernel<<<pk_blocks, NTH, 0, stream>>>(pp, wp);

  const size_t shmem = (size_t)SMEM_WORDS * sizeof(float);
  (void)hipFuncSetAttribute((const void*)classifier_fused_kernel,
                            hipFuncAttributeMaxDynamicSharedMemorySize,
                            (int)shmem);
  classifier_fused_kernel<<<BATCH, NTH, shmem, stream>>>(pp, wp, (float*)d_out);
}
